// ECCLoss_9509057593861
// MI455X (gfx1250) — matrix operand order repaired and checked by reading, not compiled, not run on hardware
//
#include <hip/hip_runtime.h>
#include <hip/hip_bf16.h>

#define NC 8192   // classes
#define ND 1024   // feature dim
#define NB 1024   // batch
#define EPSF 1e-8f

typedef __attribute__((ext_vector_type(16))) _Float16 v16h;
typedef __attribute__((ext_vector_type(8)))  float    v8f;

// order-preserving float -> uint map (for atomic min/max on floats)
__device__ __forceinline__ unsigned int mapf(float f) {
    unsigned int u = __float_as_uint(f);
    return (u & 0x80000000u) ? ~u : (u | 0x80000000u);
}
__device__ __forceinline__ float unmapf(unsigned int u) {
    unsigned int b = (u & 0x80000000u) ? (u ^ 0x80000000u) : ~u;
    return __uint_as_float(b);
}

// ---- block-wide reductions (blockDim.x == 256) ----
__device__ __forceinline__ float bsum(float v, float* red) {
    int tid = threadIdx.x;
    red[tid] = v; __syncthreads();
    for (int s = 128; s > 0; s >>= 1) { if (tid < s) red[tid] += red[tid + s]; __syncthreads(); }
    float r = red[0]; __syncthreads(); return r;
}
__device__ __forceinline__ float bmax(float v, float* red) {
    int tid = threadIdx.x;
    red[tid] = v; __syncthreads();
    for (int s = 128; s > 0; s >>= 1) { if (tid < s) red[tid] = fmaxf(red[tid], red[tid + s]); __syncthreads(); }
    float r = red[0]; __syncthreads(); return r;
}

// ---- per-class feature-table mean update + row normalize to f16 ----
__global__ void __launch_bounds__(256)
k_ft_update(const float* __restrict__ feature, const int* __restrict__ targets,
            const float* __restrict__ ft0, const float* __restrict__ cnt0,
            float* __restrict__ ftout, _Float16* __restrict__ ftn) {
    __shared__ int   tgt[NB];
    __shared__ float red[256];
    __shared__ float s_inv;
    int tid = threadIdx.x, t = blockIdx.x;
    for (int i = tid; i < NB; i += 256) tgt[i] = targets[i];
    __syncthreads();
    float acc[4] = {0.f, 0.f, 0.f, 0.f};
    int k = 0;
    for (int i = 0; i < NB; ++i) {
        if (tgt[i] == t) {
            ++k;
            const float* fr = feature + (size_t)i * ND;
            #pragma unroll
            for (int j = 0; j < 4; ++j) acc[j] += fr[tid + j * 256];
        }
    }
    const float* f0 = ft0 + (size_t)t * ND;
    float c0 = cnt0[t];
    float vals[4];
    if (k > 0) {
        float inv = 1.0f / (c0 + (float)k);
        #pragma unroll
        for (int j = 0; j < 4; ++j) vals[j] = (f0[tid + j * 256] * c0 + acc[j]) * inv;
    } else {
        #pragma unroll
        for (int j = 0; j < 4; ++j) vals[j] = f0[tid + j * 256];
    }
    float ss = 0.f;
    #pragma unroll
    for (int j = 0; j < 4; ++j) ss += vals[j] * vals[j];
    ss = bsum(ss, red);
    if (tid == 0) s_inv = 1.0f / sqrtf(ss);
    __syncthreads();
    float inv = s_inv;
    #pragma unroll
    for (int j = 0; j < 4; ++j) {
        int c = tid + j * 256;
        ftout[(size_t)t * ND + c] = vals[j];
        ftn[(size_t)t * ND + c]   = (_Float16)(vals[j] * inv);
    }
}

// ---- per-class logit-table mean update (row width 8192) ----
__global__ void __launch_bounds__(256)
k_lt_update(const float* __restrict__ logits, const int* __restrict__ targets,
            const float* __restrict__ lt0, const float* __restrict__ cnt0,
            float* __restrict__ ltout) {
    __shared__ int tgt[NB];
    int tid = threadIdx.x, t = blockIdx.x;
    for (int i = tid; i < NB; i += 256) tgt[i] = targets[i];
    __syncthreads();
    float acc[32];
    #pragma unroll
    for (int j = 0; j < 32; ++j) acc[j] = 0.f;
    int k = 0;
    for (int i = 0; i < NB; ++i) {
        if (tgt[i] == t) {
            ++k;
            const float* lr = logits + (size_t)i * NC;
            #pragma unroll
            for (int j = 0; j < 32; ++j) acc[j] += lr[tid + j * 256];
        }
    }
    const float* l0 = lt0 + (size_t)t * NC;
    float* lo = ltout + (size_t)t * NC;
    float c0 = cnt0[t];
    if (k > 0) {
        float inv = 1.0f / (c0 + (float)k);
        #pragma unroll
        for (int j = 0; j < 32; ++j) { int c = tid + j * 256; lo[c] = (l0[c] * c0 + acc[j]) * inv; }
    } else {
        #pragma unroll
        for (int j = 0; j < 32; ++j) { int c = tid + j * 256; lo[c] = l0[c]; }
    }
}

// ---- init reduction state ----
__global__ void k_init(unsigned long long* rowpack, unsigned int* gbits) {
    int i = blockIdx.x * 256 + threadIdx.x;
    if (i < NC) rowpack[i] = 0ull;
    if (i == 0) { gbits[0] = 0xFFFFFFFFu; gbits[1] = 0u; }
}

// ---- sim = ftn * ftn^T via WMMA, fused row-argmax + global min/max ----
// upper-triangular 32x32 tiles, one tile per wave, 8 waves per block.
__global__ void __launch_bounds__(256)
k_sim(const _Float16* __restrict__ ftn, unsigned long long* __restrict__ rowpack,
      unsigned int* __restrict__ gbits) {
    __shared__ float tile[8][32 * 32];   // 32 KB
    __shared__ float sred[8][64];
    int tid = threadIdx.x, wave = tid >> 5, lane = tid & 31;
    int q = blockIdx.x * 8 + wave;
    // map linear q -> (ti, tj), ti <= tj over 256x256 tile grid
    int tj = (int)((sqrtf(8.0f * (float)q + 1.0f) - 1.0f) * 0.5f);
    while ((tj + 1) * (tj + 2) / 2 <= q) ++tj;
    while (tj * (tj + 1) / 2 > q) --tj;
    int ti = q - tj * (tj + 1) / 2;
    int row0 = ti * 32, col0 = tj * 32;

    int m = lane & 15, hi = lane >> 4;
    const _Float16* arow0 = ftn + (size_t)(row0 + m) * ND;
    const _Float16* arow1 = ftn + (size_t)(row0 + 16 + m) * ND;
    const _Float16* brow0 = ftn + (size_t)(col0 + m) * ND;
    const _Float16* brow1 = ftn + (size_t)(col0 + 16 + m) * ND;

    v8f acc00 = {}, acc01 = {}, acc10 = {}, acc11 = {};
    for (int kk = 0; kk < ND; kk += 32) {
        union { uint4 u[2]; v16h h; } a0, a1, b0, b1;
        // A fragment (16x32 f16): lane holds K [hi*8, hi*8+7] and [16+hi*8, 16+hi*8+7]
        a0.u[0] = *(const uint4*)(arow0 + kk + hi * 8);
        a0.u[1] = *(const uint4*)(arow0 + kk + hi * 8 + 16);
        a1.u[0] = *(const uint4*)(arow1 + kk + hi * 8);
        a1.u[1] = *(const uint4*)(arow1 + kk + hi * 8 + 16);
        // B fragment (32x16 f16) of N^T: lane holds row (col0+m), K [hi*8, hi*8+7] and [16+hi*8, 16+hi*8+7]
        b0.u[0] = *(const uint4*)(brow0 + kk + hi * 8);
        b0.u[1] = *(const uint4*)(brow0 + kk + hi * 8 + 16);
        b1.u[0] = *(const uint4*)(brow1 + kk + hi * 8);
        b1.u[1] = *(const uint4*)(brow1 + kk + hi * 8 + 16);
        acc00 = __builtin_amdgcn_wmma_f32_16x16x32_f16(false, a0.h, false, b0.h, (short)0, acc00, false, false);
        acc01 = __builtin_amdgcn_wmma_f32_16x16x32_f16(false, a0.h, false, b1.h, (short)0, acc01, false, false);
        acc10 = __builtin_amdgcn_wmma_f32_16x16x32_f16(false, a1.h, false, b0.h, (short)0, acc10, false, false);
        acc11 = __builtin_amdgcn_wmma_f32_16x16x32_f16(false, a1.h, false, b1.h, (short)0, acc11, false, false);
    }

    // spill 32x32 tile to LDS (C/D layout: lanes 0-15 N=lane M=vgpr, lanes 16-31 N=lane-16 M=8+vgpr)
    float* tl = &tile[wave][0];
    int nl = lane & 15;
    #pragma unroll
    for (int j = 0; j < 8; ++j) {
        int ml = hi * 8 + j;
        tl[ml * 32 + nl]             = acc00[j];
        tl[ml * 32 + nl + 16]        = acc01[j];
        tl[(ml + 16) * 32 + nl]      = acc10[j];
        tl[(ml + 16) * 32 + nl + 16] = acc11[j];
    }
    __syncthreads();

    float lmin = 3.4e38f, lmax = -3.4e38f;
    {   // row scan: rows of this tile
        int r = row0 + lane;
        float bv = -3.4e38f; int bc = 0;
        for (int c = 0; c < 32; ++c) {
            float v = tl[lane * 32 + c]; int gc = col0 + c;
            lmin = fminf(lmin, v); lmax = fmaxf(lmax, v);       // global min/max includes diag
            if (gc != r && v > bv) { bv = v; bc = gc; }         // row-argmax excludes diag
        }
        atomicMax(rowpack + r, ((unsigned long long)mapf(bv) << 32) | (unsigned int)bc);
    }
    if (ti != tj) {  // symmetric contribution: transposed scan (off-diag tile only)
        int r = col0 + lane;
        float bv = -3.4e38f; int bc = 0;
        for (int c = 0; c < 32; ++c) {
            float v = tl[c * 32 + lane]; int gc = row0 + c;
            if (v > bv) { bv = v; bc = gc; }
        }
        atomicMax(rowpack + r, ((unsigned long long)mapf(bv) << 32) | (unsigned int)bc);
    }
    sred[wave][lane] = lmin; sred[wave][lane + 32] = lmax;
    __syncthreads();
    if (lane == 0) {
        float mn = 3.4e38f, mx = -3.4e38f;
        for (int l = 0; l < 32; ++l) { mn = fminf(mn, sred[wave][l]); mx = fmaxf(mx, sred[wave][l + 32]); }
        atomicMin(gbits + 0, mapf(mn));
        atomicMax(gbits + 1, mapf(mx));
    }
}

// ---- finalize similar-class tables (min-max normalization is monotonic) ----
__global__ void k_fin(const unsigned long long* __restrict__ rowpack,
                      const unsigned int* __restrict__ gbits,
                      int* __restrict__ scls, float* __restrict__ scv) {
    int r = blockIdx.x * 256 + threadIdx.x;
    if (r >= NC) return;
    float gmin = unmapf(gbits[0]), gmax = unmapf(gbits[1]);
    float inv = 1.0f / (gmax - gmin);
    unsigned long long pk = rowpack[r];
    scls[r] = (int)(pk & 0xFFFFFFFFull);
    scv[r]  = (unmapf((unsigned int)(pk >> 32)) - gmin) * inv;
}

// ---- per-sample cosine losses ----
__global__ void __launch_bounds__(256)
k_feat(const float* __restrict__ feature, const int* __restrict__ targets,
       const float* __restrict__ ftout, const int* __restrict__ scls,
       const float* __restrict__ scv, float* __restrict__ p1, float* __restrict__ p2) {
    __shared__ float red[256];
    int i = blockIdx.x, tid = threadIdx.x;
    int t = targets[i], s = scls[t];
    const float* f = feature + (size_t)i * ND;
    const float* a = ftout + (size_t)t * ND;
    const float* b = ftout + (size_t)s * ND;
    float daf = 0, dbf = 0, na = 0, nb = 0, nf = 0;
    #pragma unroll
    for (int j = 0; j < 4; ++j) {
        int c = tid + j * 256;
        float fv = f[c], av = a[c], bv = b[c];
        daf += av * fv; dbf += bv * fv; na += av * av; nb += bv * bv; nf += fv * fv;
    }
    float daf_r = bsum(daf, red), dbf_r = bsum(dbf, red);
    float na_r = bsum(na, red), nb_r = bsum(nb, red), nf_r = bsum(nf, red);
    if (tid == 0) {
        float c1 = daf_r / fmaxf(sqrtf(na_r) * sqrtf(nf_r), EPSF);
        float c2 = dbf_r / fmaxf(sqrtf(nb_r) * sqrtf(nf_r), EPSF);
        p1[i] = 1.0f - c1;
        p2[i] = c2 * scv[t];
    }
}

// ---- per-sample KL(softmax(lt[t]) || softmax(logits)) ----
__global__ void __launch_bounds__(256)
k_kl(const float* __restrict__ logits, const int* __restrict__ targets,
     const float* __restrict__ ltout, float* __restrict__ p3) {
    __shared__ float red[256];
    int i = blockIdx.x, tid = threadIdx.x, t = targets[i];
    const float* a = ltout + (size_t)t * NC;
    const float* q = logits + (size_t)i * NC;
    float ma = -3.4e38f, mq = -3.4e38f;
    for (int j = 0; j < 32; ++j) { int c = tid + j * 256; ma = fmaxf(ma, a[c]); mq = fmaxf(mq, q[c]); }
    ma = bmax(ma, red); mq = bmax(mq, red);
    float sa = 0.f, sq = 0.f;
    for (int j = 0; j < 32; ++j) { int c = tid + j * 256; sa += expf(a[c] - ma); sq += expf(q[c] - mq); }
    sa = bsum(sa, red); sq = bsum(sq, red);
    float lsa = logf(sa), lsq = logf(sq), acc = 0.f;
    for (int j = 0; j < 32; ++j) {
        int c = tid + j * 256;
        float za = a[c] - ma, zq = q[c] - mq;
        float p = expf(za) / sa;
        acc += p * ((za - lsa) - (zq - lsq));
    }
    acc = bsum(acc, red);
    if (tid == 0) p3[i] = acc;
}

// ---- fixed-order final reduction of per-sample partials ----
__global__ void k_reduce(const float* __restrict__ p1, const float* __restrict__ p2,
                         const float* __restrict__ p3, float* __restrict__ out) {
    __shared__ float red[256];
    int tid = threadIdx.x;
    float s = 0.f;
    for (int j = tid; j < NB; j += 256) s += p1[j] + p2[j];
    s = bsum(s, red);
    if (tid == 0) out[0] = s;
    float s2 = 0.f;
    for (int j = tid; j < NB; j += 256) s2 += p3[j];
    s2 = bsum(s2, red);
    if (tid == 0) out[1] = s2;
}

extern "C" void kernel_launch(void* const* d_in, const int* in_sizes, int n_in,
                              void* d_out, int out_size, void* d_ws, size_t ws_size,
                              hipStream_t stream) {
    (void)in_sizes; (void)n_in; (void)out_size; (void)ws_size;
    const float* feature = (const float*)d_in[0];
    const float* logits  = (const float*)d_in[1];
    const int*   targets = (const int*)d_in[2];
    const float* ft0     = (const float*)d_in[3];
    const float* lt0     = (const float*)d_in[4];
    const float* cnt0    = (const float*)d_in[5];

    float* out    = (float*)d_out;
    float* out_ft = out + 2;
    float* out_lt = out + 2 + (size_t)NC * ND;

    char* ws = (char*)d_ws;
    _Float16* ftn               = (_Float16*)ws;                               // 16 MB
    unsigned long long* rowpack = (unsigned long long*)(ws + (size_t)NC * ND * 2);
    unsigned int* gbits         = (unsigned int*)((char*)rowpack + (size_t)NC * 8);
    int*   scls = (int*)((char*)gbits + 256);
    float* scv  = (float*)((char*)scls + (size_t)NC * 4);
    float* p1   = (float*)((char*)scv + (size_t)NC * 4);
    float* p2   = p1 + NB;
    float* p3   = p2 + NB;

    k_init<<<(NC + 255) / 256, 256, 0, stream>>>(rowpack, gbits);
    k_ft_update<<<NC, 256, 0, stream>>>(feature, targets, ft0, cnt0, out_ft, ftn);
    k_lt_update<<<NC, 256, 0, stream>>>(logits, targets, lt0, cnt0, out_lt);
    k_sim<<<(256 * 257 / 2) / 8, 256, 0, stream>>>(ftn, rowpack, gbits);   // 4112 blocks, 8 tiles each
    k_fin<<<(NC + 255) / 256, 256, 0, stream>>>(rowpack, gbits, scls, scv);
    k_feat<<<NB, 256, 0, stream>>>(feature, targets, out_ft, scls, scv, p1, p2);
    k_kl<<<NB, 256, 0, stream>>>(logits, targets, out_lt, p3);
    k_reduce<<<1, 256, 0, stream>>>(p1, p2, p3, out);
}
